// CRDLoss_72009421684966
// MI455X (gfx1250) — matrix operand order repaired and checked by reading, not compiled, not run on hardware
//
#include <hip/hip_runtime.h>
#include <math.h>

typedef __attribute__((ext_vector_type(16))) __bf16    v16bf;
typedef __attribute__((ext_vector_type(2)))  __bf16    bf2;
typedef __attribute__((ext_vector_type(8)))  float     v8f;
typedef __attribute__((ext_vector_type(8)))  unsigned  v8u;
typedef __attribute__((ext_vector_type(4)))  unsigned  u4v;
typedef __attribute__((ext_vector_type(4)))  float     f4v;

#define DIN    1024
#define DOUT   128
#define NROWS  16384            // B*TS = 16*1024
#define KTOP   10
#define SROWS  11               // only student rows 0..10 are ever referenced
#define TINV   (1.0f / 0.07f)
#define EPSC   1e-7f
#define MTILE  64               // rows per block in the fused GEMM kernel
#define NBLK   (NROWS / MTILE)  // 256 blocks per head
#define KSTEPS (DIN / 32)       // 32 K-steps of 32
#define WSLICE (16 * DOUT)      // 2048 dwords: one packed-W K-slice
#define CPAD   20               // padded LDS row stride (dwords) per output col
#define WPKSZ  (KSTEPS * WSLICE) // 65536 dwords per head

struct BFrag { u4v lo, hi; };   // 32B, trivially copyable -> bit_cast to v16bf

#if defined(__has_builtin)
#  if __has_builtin(__builtin_amdgcn_cvt_pk_bf16_f32)
#    define HAVE_CVT_PK_BF16 1
#  endif
#  if __has_builtin(__builtin_amdgcn_s_wait_tensorcnt)
#    define HAVE_WAIT_TENSOR 1
#  endif
#endif

#ifdef HAVE_WAIT_TENSOR
#  define WAIT_TENSORCNT(n) __builtin_amdgcn_s_wait_tensorcnt(n)
#else
#  define WAIT_TENSORCNT(n) asm volatile("s_wait_tensorcnt %0" :: "i"(n) : "memory")
#endif

// ---- f32 pair -> packed bf16 (round-to-nearest-even) ------------------------
// Fallback: RNE-round both words, then merge the two high-halves with a single
// v_perm_b32 (bytes {hi[3],hi[2],lo[3],lo[2]} -> selector 0x07060302).
__device__ __forceinline__ unsigned pk2bf(float lo, float hi) {
#ifdef HAVE_CVT_PK_BF16
  bf2 p = __builtin_amdgcn_cvt_pk_bf16_f32(lo, hi);
  return __builtin_bit_cast(unsigned, p);
#else
  unsigned ul = __builtin_bit_cast(unsigned, lo);
  unsigned uh = __builtin_bit_cast(unsigned, hi);
  ul += 0x7FFFu + ((ul >> 16) & 1u);
  uh += 0x7FFFu + ((uh >> 16) & 1u);
  return __builtin_amdgcn_perm(uh, ul, 0x07060302u);
#endif
}

// =============================================================================
// Kernel 0: pack W (1024x128 f32, row-major) into bf16 B-fragment order:
//   word[kk*2048 + c*16 + p] = { bf16(W[kk*32+2p][c]), bf16(W[kk*32+2p+1][c]) }
// Each K-slice is a contiguous 8KB block; each B fragment is two 4-dword runs of
// LDS dwords after the TDM's padded copy.
// =============================================================================
__global__ __launch_bounds__(256) void pack_w_kernel(
    const float* __restrict__ W, unsigned* __restrict__ Wpk) {
  const int i  = blockIdx.x * 256 + threadIdx.x;   // 0..65535
  const int kk = i >> 11;
  const int c  = (i >> 4) & 127;
  const int p  = i & 15;
  const int k0 = kk * 32 + 2 * p;
  Wpk[i] = pk2bf(W[(size_t)k0 * DOUT + c], W[(size_t)(k0 + 1) * DOUT + c]);
}

// =============================================================================
// Kernel 1: student head — normalize(X[0:11] @ W + b). One block per side.
// =============================================================================
__global__ __launch_bounds__(256) void student_embed_kernel(
    const float* __restrict__ X, const float* __restrict__ W,
    const float* __restrict__ bias, float* __restrict__ srows_out) {
  __shared__ float y[SROWS * DOUT];
  __shared__ float inv_n[SROWS];
  const int tid = threadIdx.x;

  for (int i = tid; i < SROWS * DOUT; i += 256) {
    const int r = i >> 7, c = i & 127;
    float acc = bias[c];
    const float* xr = X + (size_t)r * DIN;
    for (int k = 0; k < DIN; ++k) acc += xr[k] * W[k * DOUT + c];
    y[i] = acc;
  }
  __syncthreads();
  if (tid < SROWS) {
    float ss = 0.0f;
    for (int c = 0; c < DOUT; ++c) { float v = y[tid * DOUT + c]; ss += v * v; }
    inv_n[tid] = 1.0f / sqrtf(ss);
  }
  __syncthreads();
  for (int i = tid; i < SROWS * DOUT; i += 256)
    srows_out[i] = y[i] * inv_n[i >> 7];
}

// =============================================================================
// TDM: DMA one packed-W slice (2048 dwords) global -> LDS with hardware
// padding: 4 pad dwords after every 16 data dwords -> LDS layout [c*CPAD + p].
//   D# group0: count=1 | lds_addr | 57-bit global_addr | type=2 ("image")
//   D# group1: data_size=4B, pad_enable, pad_interval=16dw, pad_amount=4dw,
//              1-D tile: tensor_dim0 = tile_dim0 = 2048, dim1 = 1
// =============================================================================
__device__ __forceinline__ void tdm_load_slice(const unsigned* __restrict__ gsrc,
                                               unsigned* ldsdst) {
  const unsigned long long ga = (unsigned long long)(size_t)gsrc;
  u4v g0;
  g0[0] = 1u;                                   // count = 1 (valid user D#)
  g0[1] = (unsigned)(size_t)ldsdst;             // lds_addr (bytes)
  g0[2] = (unsigned)ga;                         // global_addr[31:0]
  g0[3] = (unsigned)((ga >> 32) & 0x01FFFFFFull) | (2u << 30); // ga[56:32]|type=2
  v8u g1;
  g1[0] = (2u << 16)        // data_size = 4 bytes
        | (1u << 20)        // pad_enable
        | (3u << 22)        // pad_interval: 16 dwords between pads
        | (3u << 25);       // pad_amount:   4 dwords of pad
  g1[1] = (WSLICE & 0xFFFFu) << 16;             // tensor_dim0[15:0] = 2048
  g1[2] = (1u << 16);                           // tensor_dim1 = 1
  g1[3] = ((unsigned)WSLICE) << 16;             // tile_dim0 = 2048
  g1[4] = 0u;                                   // tile_dim1/2 unused
  g1[5] = (unsigned)WSLICE;                     // tensor_dim0_stride = 2048
  g1[6] = 0u;
  g1[7] = 0u;
  asm volatile("tensor_load_to_lds %0, %1" :: "s"(g0), "s"(g1) : "memory");
}

// =============================================================================
// Kernel 2: fused teacher embedding + contrastive loss partial.
//   64 rows (4 waves x 16) x 128 cols, K=1024 in 32 steps, bf16 WMMA f32-acc.
//   W slices double-buffered in LDS via TDM (TENSORcnt); student rows staged
//   via async global->LDS (ASYNCcnt).
// =============================================================================
__global__ __launch_bounds__(128) void embed_contrast_kernel(
    const float* __restrict__ X, const unsigned* __restrict__ Wpk,
    const float* __restrict__ bias, const float* __restrict__ srows,
    const int* __restrict__ Mp, float* __restrict__ partial_out) {
  __shared__ unsigned sWp[2][DOUT * CPAD];   // 2 x 10 KB packed-W slices
  __shared__ float sY[MTILE * DOUT];         // 32 KB un-normalized Y (+bias)
  __shared__ float sS[SROWS * DOUT];         // 5.5 KB normalized student rows
  __shared__ float sRed[128];

  const int tid   = threadIdx.x;
  const int lane  = tid & 31;
  const int wave  = tid >> 5;      // 0..3
  const int half  = lane >> 4;     // 0/1 (lane group per ISA fragment layouts)
  const int ln    = lane & 15;     // 0..15
  const int mbase = blockIdx.x * MTILE;

  // async-stage the 11 normalized student rows (1408 dwords = 352 x b128)
#pragma unroll
  for (int q = 0; q < 3; ++q) {
    const int j = q * 128 + tid;
    if (j < (SROWS * DOUT) / 4) {
      const unsigned long long ga = (unsigned long long)(size_t)(srows + j * 4);
      const unsigned la = (unsigned)(size_t)&sS[j * 4];
      asm volatile("global_load_async_to_lds_b128 %0, %1, off"
                   :: "v"(la), "v"(ga) : "memory");
    }
  }

  float biasv[8];
#pragma unroll
  for (int t = 0; t < 8; ++t) biasv[t] = bias[t * 16 + ln];

  v8f acc[8];
#pragma unroll
  for (int t = 0; t < 8; ++t)
#pragma unroll
    for (int e = 0; e < 8; ++e) acc[t][e] = 0.0f;

  // A-matrix (16x32 bf16): lanes 0-15 and 16-31 both hold M = ln.
  const float* xrow = X + (size_t)(mbase + wave * 16 + ln) * DIN;

  // prologue: TDM slice 0 -> buffer 0 (one wave drives the DMA)
  if (wave == 0) tdm_load_slice(Wpk, sWp[0]);

  for (int kk = 0; kk < KSTEPS; ++kk) {
    const int kb = kk * 32;
    const unsigned* cur = sWp[kk & 1];

    __syncthreads();   // everyone done reading the buffer the TDM will refill
    if (wave == 0) {
      if (kk + 1 < KSTEPS) {
        tdm_load_slice(Wpk + (kk + 1) * WSLICE, sWp[(kk + 1) & 1]);
        WAIT_TENSORCNT(1);   // TDM is in-order: current slice complete
      } else {
        WAIT_TENSORCNT(0);
      }
    }
    __syncthreads();   // current slice visible to all waves
    if (kk + 1 < KSTEPS) __builtin_prefetch(xrow + kb + 32, 0, 0);

    // --- A fragment: per ISA 16-bit 16x32 layout ---
    //   VGPR 0-3: K = half*8 + {0..7};  VGPR 4-7: K = 16 + half*8 + {0..7}
    const f4v f0 = *(const f4v*)(xrow + kb + half * 8);
    const f4v f1 = *(const f4v*)(xrow + kb + half * 8 + 4);
    const f4v f2 = *(const f4v*)(xrow + kb + 16 + half * 8);
    const f4v f3 = *(const f4v*)(xrow + kb + 16 + half * 8 + 4);
    v8u au;
    au[0] = pk2bf(f0.x, f0.y);  au[1] = pk2bf(f0.z, f0.w);
    au[2] = pk2bf(f1.x, f1.y);  au[3] = pk2bf(f1.z, f1.w);
    au[4] = pk2bf(f2.x, f2.y);  au[5] = pk2bf(f2.z, f2.w);
    au[6] = pk2bf(f3.x, f3.y);  au[7] = pk2bf(f3.z, f3.w);
    const v16bf a = __builtin_bit_cast(v16bf, au);

    // --- B fragments: dwords half*4..+3 and 8+half*4..+3 in LDS -> two ds_load_b128 ---
#pragma unroll
    for (int t = 0; t < 8; ++t) {
      const int cb = (t * 16 + ln) * CPAD + half * 4;
      BFrag bw;
      bw.lo = *(const u4v*)&cur[cb];
      bw.hi = *(const u4v*)&cur[cb + 8];
      const v16bf b = __builtin_bit_cast(v16bf, bw);
      acc[t] = __builtin_amdgcn_wmma_f32_16x16x32_bf16(
          false, a, false, b, (short)0, acc[t], false, false);
    }
  }

  // C/D layout: VGPR v holds row half*8+v, col = t*16 + ln.
#pragma unroll
  for (int t = 0; t < 8; ++t)
#pragma unroll
    for (int v = 0; v < 8; ++v)
      sY[(wave * 16 + half * 8 + v) * DOUT + t * 16 + ln] = acc[t][v] + biasv[t];

  // student rows must have landed before the loss phase
  asm volatile("s_wait_asynccnt 0x0" ::: "memory");
  __syncthreads();

  const float mPn = 9.0f / (float)(*Mp);   // m * Pn, m = KTOP-1 = 9
  float local = 0.0f;
  if (tid < MTILE) {
    const float* yrow = &sY[tid * DOUT];
    float ss = 0.0f;
    for (int c = 0; c < DOUT; ++c) ss += yrow[c] * yrow[c];
    const float inv_norm = 1.0f / sqrtf(ss);
    const int n = mbase + tid;
    float lsum = 0.0f;
#pragma unroll
    for (int j = 0; j < KTOP; ++j) {
      const int sidx = j + ((j >= n) ? 1 : 0);   // skip self among rows 0..10
      const float* srow = &sS[sidx * DOUT];
      float dot = 0.0f;
      for (int c = 0; c < DOUT; ++c) dot += yrow[c] * srow[c];
      const float P = expf(dot * inv_norm * TINV);
      lsum += (j == 0) ? logf(P / (P + mPn + EPSC))
                       : logf(mPn / (P + mPn + EPSC));
    }
    local = -lsum * (1.0f / (float)NROWS);
  }
  sRed[tid] = local;
  __syncthreads();
  for (int s = 64; s > 0; s >>= 1) {
    if (tid < s) sRed[tid] += sRed[tid + s];
    __syncthreads();
  }
  if (tid == 0) partial_out[blockIdx.x] = sRed[0];
}

// =============================================================================
// Kernel 3: deterministic final reduction of 4 x NBLK partials -> 2 losses.
// =============================================================================
__global__ __launch_bounds__(256) void final_reduce_kernel(
    const float* __restrict__ partials, float* __restrict__ out) {
  __shared__ float red[256];
  const int tid = threadIdx.x;

  red[tid] = partials[tid] + partials[NBLK + tid];                // entity E+R
  __syncthreads();
  for (int s = 128; s > 0; s >>= 1) {
    if (tid < s) red[tid] += red[tid + s];
    __syncthreads();
  }
  if (tid == 0) out[0] = red[0];
  __syncthreads();

  red[tid] = partials[2 * NBLK + tid] + partials[3 * NBLK + tid]; // rel E+R
  __syncthreads();
  for (int s = 128; s > 0; s >>= 1) {
    if (tid < s) red[tid] += red[tid + s];
    __syncthreads();
  }
  if (tid == 0) out[1] = red[0];
}

// =============================================================================
extern "C" void kernel_launch(void* const* d_in, const int* in_sizes, int n_in,
                              void* d_out, int out_size, void* d_ws, size_t ws_size,
                              hipStream_t stream) {
  (void)in_sizes; (void)n_in; (void)out_size; (void)ws_size;
  // setup_inputs() dict order:
  const float* ent_s  = (const float*)d_in[0];
  const float* rel_s  = (const float*)d_in[1];
  const float* ent_tE = (const float*)d_in[2];
  const float* rel_tE = (const float*)d_in[3];
  const float* ent_tR = (const float*)d_in[4];
  const float* rel_tR = (const float*)d_in[5];
  // d_in[6..9] logits: unused by reference
  const float* We_s  = (const float*)d_in[10];
  const float* be_s  = (const float*)d_in[11];
  const float* We_tE = (const float*)d_in[12];
  const float* be_tE = (const float*)d_in[13];
  const float* We_tR = (const float*)d_in[14];
  const float* be_tR = (const float*)d_in[15];
  const float* Wr_s  = (const float*)d_in[16];
  const float* br_s  = (const float*)d_in[17];
  const float* Wr_tE = (const float*)d_in[18];
  const float* br_tE = (const float*)d_in[19];
  const float* Wr_tR = (const float*)d_in[20];
  const float* br_tR = (const float*)d_in[21];
  // d_in[22..23] contrast_idx/idx: unused by reference
  const int*   Mp    = (const int*)d_in[24];

  float* out   = (float*)d_out;
  float* s_ent = (float*)d_ws;                 // 11*128 floats
  float* s_rel = s_ent + SROWS * DOUT;         // 11*128 floats
  float* parts = s_rel + SROWS * DOUT;         // 4*NBLK floats
  unsigned* wpk = (unsigned*)(parts + 4 * NBLK); // 16B-aligned; 4 x WPKSZ u32

  // 0) pack the four teacher weight matrices to bf16 fragment order
  pack_w_kernel<<<WPKSZ / 256, 256, 0, stream>>>(We_tE, wpk + 0 * WPKSZ);
  pack_w_kernel<<<WPKSZ / 256, 256, 0, stream>>>(We_tR, wpk + 1 * WPKSZ);
  pack_w_kernel<<<WPKSZ / 256, 256, 0, stream>>>(Wr_tE, wpk + 2 * WPKSZ);
  pack_w_kernel<<<WPKSZ / 256, 256, 0, stream>>>(Wr_tR, wpk + 3 * WPKSZ);

  // 1) student heads (only rows 0..10 are ever used)
  student_embed_kernel<<<1, 256, 0, stream>>>(ent_s, We_s, be_s, s_ent);
  student_embed_kernel<<<1, 256, 0, stream>>>(rel_s, Wr_s, br_s, s_rel);

  // 2) four fused teacher-embedding + loss passes
  embed_contrast_kernel<<<NBLK, 128, 0, stream>>>(ent_tE, wpk + 0 * WPKSZ, be_tE, s_ent, Mp, parts + 0 * NBLK);
  embed_contrast_kernel<<<NBLK, 128, 0, stream>>>(ent_tR, wpk + 1 * WPKSZ, be_tR, s_ent, Mp, parts + 1 * NBLK);
  embed_contrast_kernel<<<NBLK, 128, 0, stream>>>(rel_tE, wpk + 2 * WPKSZ, br_tE, s_rel, Mp, parts + 2 * NBLK);
  embed_contrast_kernel<<<NBLK, 128, 0, stream>>>(rel_tR, wpk + 3 * WPKSZ, br_tR, s_rel, Mp, parts + 3 * NBLK);

  // 3) deterministic final reduction -> (entity_s_loss, rel_s_loss)
  final_reduce_kernel<<<1, 256, 0, stream>>>(parts, out);
}
